// MultiHeadAttention_54099408060546
// MI455X (gfx1250) — matrix operand order repaired and checked by reading, not compiled, not run on hardware
//
#include <hip/hip_runtime.h>

// ---------------------------------------------------------------------------
// CDNA5 (gfx1250) multi-head attention forward.
//  - all matmuls: v_wmma_f32_16x16x32_bf16 (wave32), f32 accumulate
//  - operands pre-converted to bf16 so GEMM tile fills are byte copies
//  - tile fills via Tensor Data Mover (tensor_load_to_lds + s_wait_tensorcnt),
//    double buffered; sync fallback if the builtin is unavailable
// ---------------------------------------------------------------------------

typedef __attribute__((ext_vector_type(4)))  unsigned int v4u;
typedef __attribute__((ext_vector_type(8)))  unsigned int v8u;
typedef __attribute__((ext_vector_type(16))) __bf16      v16bf;
typedef __attribute__((ext_vector_type(8)))  float       v8f;
typedef __attribute__((ext_vector_type(4)))  unsigned int u32x4;
typedef __attribute__((ext_vector_type(8)))  int          i32x8;
typedef __attribute__((ext_vector_type(4)))  int          i32x4;

#define LDSPAD 40   // 32 bf16 data + 8 pad = 80 B row pitch (16B aligned)

#if defined(__HIP_DEVICE_COMPILE__) && \
    __has_builtin(__builtin_amdgcn_tensor_load_to_lds) && \
    __has_builtin(__builtin_amdgcn_s_wait_tensorcnt)
#define USE_TDM 1
#else
#define USE_TDM 0
#endif

__device__ __forceinline__ unsigned short f2bf(float f) {
    unsigned int u = __builtin_bit_cast(unsigned int, f);
    u += 0x7FFFu + ((u >> 16) & 1u);
    return (unsigned short)(u >> 16);
}

// pack two f32 -> bf16x2 (v_cvt_pk_bf16_f32 when available)
__device__ __forceinline__ unsigned int pk2bf(float a, float b) {
#if defined(__HIP_DEVICE_COMPILE__) && __has_builtin(__builtin_amdgcn_cvt_pk_bf16_f32)
    typedef __attribute__((ext_vector_type(2))) __bf16 v2bf;
    v2bf r = __builtin_amdgcn_cvt_pk_bf16_f32(a, b);
    return __builtin_bit_cast(unsigned int, r);
#else
    return (unsigned int)f2bf(a) | ((unsigned int)f2bf(b) << 16);
#endif
}

__device__ __forceinline__ v16bf frag_cast(v4u lo, v4u hi) {
    union { v8u u; v16bf b; } cv;
    cv.u[0] = lo[0]; cv.u[1] = lo[1]; cv.u[2] = lo[2]; cv.u[3] = lo[3];
    cv.u[4] = hi[0]; cv.u[5] = hi[1]; cv.u[6] = hi[2]; cv.u[7] = hi[3];
    return cv.b;
}

#if USE_TDM
// LDS byte offset of a __shared__ object (generic -> AS(3) -> int)
__device__ __forceinline__ unsigned lds_off(const void* p) {
    return (unsigned)(unsigned long long)
        (__attribute__((address_space(3))) const void*)p;
}

// 2D bf16 tile DMA: tile_d0 x tile_d1 elements, row stride `stride0` elems,
// LDS padded to LDSPAD (pad_enable: 16 DW data + 4 DW pad per row).
__device__ __forceinline__ void tdm_load_2d(unsigned lds_addr,
                                            const unsigned short* gptr,
                                            unsigned tile_d0, unsigned tile_d1,
                                            unsigned long long tensor_d0,
                                            unsigned long long tensor_d1,
                                            unsigned long long stride0) {
    unsigned long long ga = (unsigned long long)gptr;
    u32x4 g0;
    g0[0] = 1u;                                   // count=1, user descriptor
    g0[1] = lds_addr;                             // lds_addr[31:0]
    g0[2] = (unsigned)(ga & 0xFFFFFFFFu);         // global_addr[31:0]
    g0[3] = (unsigned)((ga >> 32) & 0x1FFFFFFu)   // global_addr[56:32]
          | 0x80000000u;                          // type=2 (image)
    i32x8 g1;
    g1[0] = (int)((1u << 16)      // data_size = 2 bytes
                | (1u << 20)      // pad_enable
                | (3u << 22)      // pad_interval: 16 DWORDs
                | (3u << 25));    // pad_amount:   4 DWORDs
    g1[1] = (int)((tensor_d0 & 0xFFFFu) << 16);
    g1[2] = (int)(((tensor_d0 >> 16) & 0xFFFFu) | ((tensor_d1 & 0xFFFFu) << 16));
    g1[3] = (int)(((tensor_d1 >> 16) & 0xFFFFu) | (tile_d0 << 16));
    g1[4] = (int)(tile_d1 & 0xFFFFu);             // tile_dim2 = 0
    g1[5] = (int)(stride0 & 0xFFFFFFFFu);
    g1[6] = (int)((stride0 >> 32) & 0xFFFFu);     // tensor_dim1_stride = 0
    g1[7] = 0;
    i32x4 z4 = {0, 0, 0, 0};
    i32x8 z8 = {0, 0, 0, 0, 0, 0, 0, 0};
    // 6-arg form (clang-23 / therock-10.0 headers)
    __builtin_amdgcn_tensor_load_to_lds(g0, g1, z4, z4, z8, 0);
}
#endif

// One 32-deep WMMA step: wave computes 2x4 16x16 tiles from padded LDS.
// Fragment layouts per CDNA5 ISA 7.12.2 (wave32).
__device__ __forceinline__ void mm_step(const unsigned short (*As)[LDSPAD],
                                        const unsigned short (*Bs)[LDSPAD],
                                        int lane, int wm, int wn, v8f acc[2][4]) {
    const int arow = lane & 15;
    const int akh  = (lane < 16) ? 0 : 8;   // A: K {0..7,16..23} / {8..15,24..31}
    v16bf af[2];
#pragma unroll
    for (int ms = 0; ms < 2; ++ms) {
        const unsigned short* ap = &As[wm + ms * 16 + arow][0];
        v4u lo = *(const v4u*)(ap + akh);
        v4u hi = *(const v4u*)(ap + 16 + akh);
        af[ms] = frag_cast(lo, hi);
    }
    const int bkh = (lane < 16) ? 0 : 8;    // B: K {0..7,16..23} / {8..15,24..31}, col = lane&15
#pragma unroll
    for (int ns = 0; ns < 4; ++ns) {
        const unsigned short* bp = &Bs[wn + ns * 16 + (lane & 15)][0];
        v4u lo = *(const v4u*)(bp + bkh);
        v4u hi = *(const v4u*)(bp + 16 + bkh);
        v16bf bf = frag_cast(lo, hi);
        acc[0][ns] = __builtin_amdgcn_wmma_f32_16x16x32_bf16(
            false, af[0], false, bf, (short)0, acc[0][ns], false, false);
        acc[1][ns] = __builtin_amdgcn_wmma_f32_16x16x32_bf16(
            false, af[1], false, bf, (short)0, acc[1][ns], false, false);
    }
}

// Batched GEMM, bf16 x bf16 -> f32 acc. A: [M][K] (stride lda), B: [N][K]
// (stride ldb, i.e. pre-transposed), C: f32 or bf16. Tile fills via TDM.
// Requires M%128==0, N%128==0, K%32==0 (all call sites satisfy this).
__global__ void __launch_bounds__(256)
gemm_bf16_tdm(const unsigned short* __restrict__ A,
              const unsigned short* __restrict__ Bm,
              const float* __restrict__ bias,
              void* __restrict__ Cv, int cIsBf16,
              int M, int N, int K, int lda, int ldb, int ldc,
              int Hdiv,
              long long sAb, long long sAh, long long sBb, long long sBh,
              long long sCb, long long sCh, float scale)
{
    __shared__ unsigned short As[2][128][LDSPAD];
    __shared__ unsigned short Bs[2][128][LDSPAD];

    const int tid  = threadIdx.x;
    const int lane = tid & 31;
    const int wave = tid >> 5;
    const int wm   = (wave & 3) * 32;
    const int wn   = (wave >> 2) * 64;
    const int m0 = blockIdx.y * 128;
    const int n0 = blockIdx.x * 128;

    const int z  = blockIdx.z;
    const int bz = z / Hdiv;
    const int hz = z % Hdiv;
    const unsigned short* Ab = A  + bz * sAb + hz * sAh;
    const unsigned short* Bb = Bm + bz * sBb + hz * sBh;
    const long long offC = bz * sCb + hz * sCh;

    v8f acc[2][4] = {};

    auto issue = [&](int p, int k0) {
#if USE_TDM
        if (wave == 0) {
            tdm_load_2d(lds_off(&As[p][0][0]),
                        Ab + (long long)m0 * lda + k0, 32u, 128u,
                        (unsigned long long)(K - k0),
                        (unsigned long long)(M - m0),
                        (unsigned long long)lda);
            tdm_load_2d(lds_off(&Bs[p][0][0]),
                        Bb + (long long)n0 * ldb + k0, 32u, 128u,
                        (unsigned long long)(K - k0),
                        (unsigned long long)(N - n0),
                        (unsigned long long)ldb);
        }
#else
        // sync fallback: pure 16B copies (dims are tile-aligned at call sites)
#pragma unroll
        for (int i = 0; i < 2; ++i) {
            int linear = tid + i * 256;          // 0..511
            int row = linear >> 2;               // 0..127
            int c8  = (linear & 3) << 3;         // 0,8,16,24
            v4u a = *(const v4u*)(Ab + (long long)(m0 + row) * lda + k0 + c8);
            *(v4u*)&As[p][row][c8] = a;
            v4u b = *(const v4u*)(Bb + (long long)(n0 + row) * ldb + k0 + c8);
            *(v4u*)&Bs[p][row][c8] = b;
        }
#endif
    };

    int p = 0;
    issue(0, 0);
    for (int k0 = 0; k0 < K; k0 += 32) {
#if USE_TDM
        if (wave == 0) __builtin_amdgcn_s_wait_tensorcnt(0);
#endif
        __syncthreads();                      // tiles in buffer p are ready
        if (k0 + 32 < K) issue(1 - p, k0 + 32);
        mm_step(As[p], Bs[p], lane, wm, wn, acc);
        __syncthreads();                      // all waves done reading p
        p ^= 1;
    }

    // epilogue: VGPR i -> M = i (+8 for lanes 16..31), N = lane&15
    const int rbase = m0 + wm + ((lane < 16) ? 0 : 8);
    if (cIsBf16) {
        unsigned short* Cb = (unsigned short*)Cv + offC;
#pragma unroll
        for (int ns = 0; ns < 4; ++ns) {
            int col = n0 + wn + ns * 16 + (lane & 15);
            float bv = (bias != nullptr) ? bias[col] : 0.f;
#pragma unroll
            for (int ms = 0; ms < 2; ++ms)
#pragma unroll
                for (int i = 0; i < 8; ++i) {
                    int row = rbase + ms * 16 + i;
                    Cb[(long long)row * ldc + col] =
                        f2bf(acc[ms][ns][i] * scale + bv);
                }
        }
    } else {
        float* Cb = (float*)Cv + offC;
#pragma unroll
        for (int ns = 0; ns < 4; ++ns) {
            int col = n0 + wn + ns * 16 + (lane & 15);
            float bv = (bias != nullptr) ? bias[col] : 0.f;
#pragma unroll
            for (int ms = 0; ms < 2; ++ms)
#pragma unroll
                for (int i = 0; i < 8; ++i) {
                    int row = rbase + ms * 16 + i;
                    Cb[(long long)row * ldc + col] = acc[ms][ns][i] * scale + bv;
                }
        }
    }
}

// attn(f32 [M][K]) @ V(bf16 [K][N], stride ldb) -> bf16 C. N=64 here, so the
// right half of the 128-wide tile is zero-filled and stores are guarded.
__global__ void __launch_bounds__(256)
gemm_attnv(const float* __restrict__ A, const unsigned short* __restrict__ Bm,
           unsigned short* __restrict__ C,
           int M, int N, int K, int lda, int ldb, int ldc, int Hdiv,
           long long sAb, long long sAh, long long sBb, long long sBh,
           long long sCb, long long sCh)
{
    __shared__ unsigned short As[128][LDSPAD];
    __shared__ unsigned short Bs[128][LDSPAD];

    const int tid  = threadIdx.x;
    const int lane = tid & 31;
    const int wave = tid >> 5;
    const int wm   = (wave & 3) * 32;
    const int wn   = (wave >> 2) * 64;
    const int m0 = blockIdx.y * 128;
    const int n0 = blockIdx.x * 128;

    const int z  = blockIdx.z;
    const int bz = z / Hdiv;
    const int hz = z % Hdiv;
    const float*          Ab = A  + bz * sAb + hz * sAh;
    const unsigned short* Bb = Bm + bz * sBb + hz * sBh;
    unsigned short*       Cb = C  + bz * sCb + hz * sCh;

    v8f acc[2][4] = {};

    for (int k0 = 0; k0 < K; k0 += 32) {
        __syncthreads();
        // A tile: f32 -> bf16 (M,K tile-aligned for attn)
#pragma unroll
        for (int i = 0; i < 4; ++i) {
            int linear = tid + i * 256;
            int row = linear >> 3;
            int c4  = (linear & 7) << 2;
            float4 val = *(const float4*)(Ab + (long long)(m0 + row) * lda + k0 + c4);
            uint2 pk;
            pk.x = pk2bf(val.x, val.y);
            pk.y = pk2bf(val.z, val.w);
            *(uint2*)&As[row][c4] = pk;
        }
        // B tile: bf16 [K][N] -> LDS [n][k] (transpose gather, guarded on N)
#pragma unroll
        for (int i = 0; i < 4; ++i) {
            int linear = tid + i * 256;
            int nl = linear & 127;
            int kq = (linear >> 7) << 2;
            int gn = n0 + nl;
            unsigned short e[4];
#pragma unroll
            for (int j = 0; j < 4; ++j) {
                int gk = k0 + kq + j;
                e[j] = (gn < N) ? Bb[(long long)gk * ldb + gn] : (unsigned short)0;
            }
            uint2 pk;
            pk.x = (unsigned)e[0] | ((unsigned)e[1] << 16);
            pk.y = (unsigned)e[2] | ((unsigned)e[3] << 16);
            *(uint2*)&Bs[nl][kq] = pk;
        }
        __syncthreads();
        mm_step(As, Bs, lane, wm, wn, acc);
    }

    const int rbase = m0 + wm + ((lane < 16) ? 0 : 8);
#pragma unroll
    for (int ns = 0; ns < 4; ++ns) {
        int col = n0 + wn + ns * 16 + (lane & 15);
#pragma unroll
        for (int ms = 0; ms < 2; ++ms)
#pragma unroll
            for (int i = 0; i < 8; ++i) {
                int row = rbase + ms * 16 + i;
                if (row < M && col < N)
                    Cb[(long long)row * ldc + col] = f2bf(acc[ms][ns][i]);
            }
    }
}

// streaming f32 -> bf16 (n multiple of 4)
__global__ void __launch_bounds__(256)
cvt_f32_to_bf16(const float* __restrict__ s, unsigned short* __restrict__ d,
                long long n)
{
    long long i0 = ((long long)blockIdx.x * 256 + threadIdx.x) * 4;
    if (i0 + 3 < n) {
        float4 v = *(const float4*)(s + i0);
        uint2 pk;
        pk.x = pk2bf(v.x, v.y);
        pk.y = pk2bf(v.z, v.w);
        *(uint2*)(d + i0) = pk;
    }
}

// W[K][N] f32 -> WT[N][K] bf16 (K,N multiples of 64)
__global__ void __launch_bounds__(256)
transpose_cvt_bf16(const float* __restrict__ W, unsigned short* __restrict__ WT,
                   int K, int N)
{
    __shared__ unsigned short T[64][72];   // [n][k], 144 B pitch (16B multiple)
    const int tid = threadIdx.x;
    const int k0 = blockIdx.y * 64, n0 = blockIdx.x * 64;
#pragma unroll
    for (int i = 0; i < 4; ++i) {
        int linear = tid + i * 256;          // 0..1023
        int kr = linear >> 4;                // 0..63
        int c4 = (linear & 15) << 2;         // 0..60
        float4 v = *(const float4*)(W + (long long)(k0 + kr) * N + n0 + c4);
        T[c4 + 0][kr] = f2bf(v.x);
        T[c4 + 1][kr] = f2bf(v.y);
        T[c4 + 2][kr] = f2bf(v.z);
        T[c4 + 3][kr] = f2bf(v.w);
    }
    __syncthreads();
#pragma unroll
    for (int i = 0; i < 2; ++i) {
        int linear = tid + i * 256;          // 0..511
        int nr = linear >> 3;                // 0..63
        int c8 = (linear & 7) << 3;          // 0..56
        v4u v = *(const v4u*)&T[nr][c8];
        *(v4u*)(WT + (long long)(n0 + nr) * K + k0 + c8) = v;
    }
}

// Row softmax over exactly 2048 columns, in place.
__global__ void __launch_bounds__(256)
softmax2048(float* __restrict__ attn)
{
    __shared__ float red[256];
    float* p = attn + (long long)blockIdx.x * 2048ll;
    const int tid = threadIdx.x;

    float v[8];
    float mx = -3.402823466e38f;
#pragma unroll
    for (int i = 0; i < 8; ++i) { v[i] = p[tid + i * 256]; mx = fmaxf(mx, v[i]); }
    red[tid] = mx;
    __syncthreads();
    for (int s = 128; s > 0; s >>= 1) {
        if (tid < s) red[tid] = fmaxf(red[tid], red[tid + s]);
        __syncthreads();
    }
    mx = red[0];
    __syncthreads();

    float sum = 0.f;
#pragma unroll
    for (int i = 0; i < 8; ++i) { v[i] = __expf(v[i] - mx); sum += v[i]; }
    red[tid] = sum;
    __syncthreads();
    for (int s = 128; s > 0; s >>= 1) {
        if (tid < s) red[tid] += red[tid + s];
        __syncthreads();
    }
    float inv = 1.0f / red[0];
#pragma unroll
    for (int i = 0; i < 8; ++i) p[tid + i * 256] = v[i] * inv;
}

extern "C" void kernel_launch(void* const* d_in, const int* in_sizes, int n_in,
                              void* d_out, int out_size, void* d_ws, size_t ws_size,
                              hipStream_t stream)
{
    (void)in_sizes; (void)n_in; (void)out_size; (void)ws_size;
    const int Bn = 2, S = 2048, D = 1024, H = 16, DK = 64;

    const float* Q_in = (const float*)d_in[0];
    const float* K_in = (const float*)d_in[1];
    const float* V_in = (const float*)d_in[2];
    const float* Wq   = (const float*)d_in[3];
    const float* bq   = (const float*)d_in[4];
    const float* Wk   = (const float*)d_in[5];
    const float* bk   = (const float*)d_in[6];
    const float* Wv   = (const float*)d_in[7];
    const float* bv   = (const float*)d_in[8];
    const float* Wo   = (const float*)d_in[9];
    const float* bo   = (const float*)d_in[10];

    float* out  = (float*)d_out;                     // [B,S,D] f32
    float* attn = out + (size_t)Bn * S * D;          // [B,H,S,S] f32

    const size_t nQ = (size_t)Bn * S * D;            // 4 Mi elements
    const size_t nW = (size_t)D * D;                 // 1 Mi elements
    unsigned short* Qc  = (unsigned short*)d_ws;     // bf16 inputs
    unsigned short* Kc  = Qc  + nQ;
    unsigned short* Vc  = Kc  + nQ;
    unsigned short* WqT = Vc  + nQ;                  // bf16 [N][K] weights
    unsigned short* WkT = WqT + nW;
    unsigned short* WvT = WkT + nW;
    unsigned short* WoT = WvT + nW;
    unsigned short* qb  = WoT + nW;                  // bf16 projections
    unsigned short* kb  = qb  + nQ;
    unsigned short* vb  = kb  + nQ;
    unsigned short* hob = Qc;   // head-concat output reuses Qc (dead by then)

    const dim3 blk(256, 1, 1);
    const int Mrows = Bn * S;                        // 4096
    const long long Z = 0;

    // 1) operand conversion to bf16
    const unsigned cvtBlocks = (unsigned)(nQ / 4 / 256);
    cvt_f32_to_bf16<<<dim3(cvtBlocks), blk, 0, stream>>>(Q_in, Qc, (long long)nQ);
    cvt_f32_to_bf16<<<dim3(cvtBlocks), blk, 0, stream>>>(K_in, Kc, (long long)nQ);
    cvt_f32_to_bf16<<<dim3(cvtBlocks), blk, 0, stream>>>(V_in, Vc, (long long)nQ);
    const dim3 gtr(D / 64, D / 64, 1);
    transpose_cvt_bf16<<<gtr, blk, 0, stream>>>(Wq, WqT, D, D);
    transpose_cvt_bf16<<<gtr, blk, 0, stream>>>(Wk, WkT, D, D);
    transpose_cvt_bf16<<<gtr, blk, 0, stream>>>(Wv, WvT, D, D);
    transpose_cvt_bf16<<<gtr, blk, 0, stream>>>(Wo, WoT, D, D);

    // 2) Q/K/V projections -> bf16
    const dim3 gproj(D / 128, Mrows / 128, 1);
    gemm_bf16_tdm<<<gproj, blk, 0, stream>>>(Qc, WqT, bq, qb, 1,
        Mrows, D, D, D, D, D, 1, Z, Z, Z, Z, Z, Z, 1.0f);
    gemm_bf16_tdm<<<gproj, blk, 0, stream>>>(Kc, WkT, bk, kb, 1,
        Mrows, D, D, D, D, D, 1, Z, Z, Z, Z, Z, Z, 1.0f);
    gemm_bf16_tdm<<<gproj, blk, 0, stream>>>(Vc, WvT, bv, vb, 1,
        Mrows, D, D, D, D, D, 1, Z, Z, Z, Z, Z, Z, 1.0f);

    // 3) scores = (q_h @ k_h^T)/8 -> f32 attn output region
    const dim3 gsc(S / 128, S / 128, Bn * H);
    gemm_bf16_tdm<<<gsc, blk, 0, stream>>>(qb, kb, nullptr, attn, 0,
        S, S, DK, D, D, S, H,
        (long long)S * D, (long long)DK,
        (long long)S * D, (long long)DK,
        (long long)H * S * S, (long long)S * S, 0.125f);

    // 4) in-place row softmax
    softmax2048<<<dim3((unsigned)(Bn * H * S)), blk, 0, stream>>>(attn);

    // 5) head outputs = attn_h @ v_h -> bf16 concat [B,S,D] (col h*DK)
    const dim3 gav(1, S / 128, Bn * H);
    gemm_attnv<<<gav, blk, 0, stream>>>(attn, vb, hob,
        S, DK, S, S, D, D, H,
        (long long)H * S * S, (long long)S * S,
        (long long)S * D, (long long)DK,
        (long long)S * D, (long long)DK);

    // 6) output projection -> f32
    gemm_bf16_tdm<<<gproj, blk, 0, stream>>>(hob, WoT, bo, out, 0,
        Mrows, D, D, D, D, D, 1, Z, Z, Z, Z, Z, Z, 1.0f);
}
